// SoftDDT_55868934586699
// MI455X (gfx1250) — compile-verified, matrix operand order checked by reading, not run on hardware
//
#include <hip/hip_runtime.h>

// ---------------- CDNA5 WMMA types ----------------
typedef __bf16 v16bf __attribute__((ext_vector_type(16)));
typedef float  v8f   __attribute__((ext_vector_type(8)));

union BFrag { v16bf v; __bf16 h[16]; uint4 q[2]; };
union AccU  { v8f v; float f[8]; };

static constexpr int kF = 256;          // features
static constexpr int kC = 256;          // classes
static constexpr int kWaves = 4;        // waves per block (wave32)
static constexpr int kRowsPerWave = 16;
static constexpr int kRowsPerBlock = kWaves * kRowsPerWave;   // 64
static constexpr int kPS = 264;         // padded LDS row stride (elements) -> bank stride 4 dwords

extern "C" __global__ void __launch_bounds__(kWaves * 32, 1)
soft_tree_fused(const float* __restrict__ X, const float* __restrict__ Wg,
                const float* __restrict__ bg, const float* __restrict__ Lg,
                float* __restrict__ out, int* __restrict__ idx_out,
                int numTiles) {
  extern __shared__ __bf16 smem[];
  __bf16* Wlds = smem;                     // [256][kPS]  W (bf16), row=node (row 255 = 0)
  __bf16* LT   = Wlds + 256 * kPS;         // [256][kPS]  leaf_probs^T (bf16), row=class
  __bf16* vals = LT   + 256 * kPS;         // [kWaves*16][kPS] sigmoid gates (bf16)

  const int tid  = threadIdx.x;
  const int wave = tid >> 5;
  const int lane = tid & 31;
  const int m    = lane & 15;   // row (A/M) or col (B/N) index within 16-tile
  const int half = lane >> 4;   // K-half selector per WMMA bf16 layout

  // ---- stage W (bf16) and leaf^T (bf16) into LDS (native v_cvt_pk_bf16_f32) ----
  for (int i = tid; i < 256 * 256; i += blockDim.x) {
    const int r = i >> 8, c = i & 255;
    const float w = (r < 255) ? Wg[r * 256 + c] : 0.0f;   // node 255 doesn't exist
    Wlds[r * kPS + c] = (__bf16)w;
    LT[c * kPS + r]   = (__bf16)Lg[i];                    // transpose: class-major
  }
  __syncthreads();

  for (int tile = blockIdx.x; tile < numTiles; tile += gridDim.x) {
    const int row0 = tile * kRowsPerBlock + wave * kRowsPerWave;
    __bf16* vw = vals + (wave * 16) * kPS;                // this wave's [16][kPS] gates
    const float* xrow = X + (size_t)(row0 + m) * kF;

    // ---- build the 8 x-A-fragments once per tile (hoisted; only depend on kk) ----
    BFrag af[8];
    #pragma unroll
    for (int kk = 0; kk < 8; ++kk) {
      const int kbase = kk * 32 + half * 8;
      const float4 q0 = ((const float4*)(xrow + kbase))[0];
      const float4 q1 = ((const float4*)(xrow + kbase))[1];
      const float4 q2 = ((const float4*)(xrow + kbase + 16))[0];
      const float4 q3 = ((const float4*)(xrow + kbase + 16))[1];
      af[kk].h[0] = (__bf16)q0.x; af[kk].h[1] = (__bf16)q0.y;
      af[kk].h[2] = (__bf16)q0.z; af[kk].h[3] = (__bf16)q0.w;
      af[kk].h[4] = (__bf16)q1.x; af[kk].h[5] = (__bf16)q1.y;
      af[kk].h[6] = (__bf16)q1.z; af[kk].h[7] = (__bf16)q1.w;
      af[kk].h[8] = (__bf16)q2.x; af[kk].h[9] = (__bf16)q2.y;
      af[kk].h[10]= (__bf16)q2.z; af[kk].h[11]= (__bf16)q2.w;
      af[kk].h[12]= (__bf16)q3.x; af[kk].h[13]= (__bf16)q3.y;
      af[kk].h[14]= (__bf16)q3.z; af[kk].h[15]= (__bf16)q3.w;
    }

    // ================= phase 1: gates = sigmoid(x @ W^T + b) =================
    for (int nt = 0; nt < 16; ++nt) {
      AccU acc;
      #pragma unroll
      for (int r = 0; r < 8; ++r) acc.f[r] = 0.0f;
      const int node = nt * 16 + m;                       // lane's output column (N)
      #pragma unroll
      for (int kk = 0; kk < 8; ++kk) {
        const int kbase = kk * 32 + half * 8;
        // B fragment: B[k][n] = W[n][k] -> contiguous 16B runs of W row `node`
        BFrag b;
        b.q[0] = ((const uint4*)(Wlds + node * kPS + kbase))[0];
        b.q[1] = ((const uint4*)(Wlds + node * kPS + kbase + 16))[0];
        acc.v = __builtin_amdgcn_wmma_f32_16x16x32_bf16(
            false, af[kk].v, false, b.v, (short)0, acc.v, false, false);
      }
      const float bb = bg[node];
      #pragma unroll
      for (int r = 0; r < 8; ++r) {   // D layout: M = r + half*8, N = m
        const float s = 1.0f / (1.0f + __expf(-(acc.f[r] + bb)));
        vw[(half * 8 + r) * kPS + node] = (__bf16)s;
      }
    }
    // per-wave LDS store->load ordering is guaranteed (DS ops in-order per wave)

    // ====== phase 2: leaf path products, built directly into A fragments ======
    const __bf16* vr = vw + m * kPS;   // this lane's row of gates
    #pragma unroll
    for (int kk = 0; kk < 8; ++kk) {
      BFrag pa;
      #pragma unroll
      for (int gsel = 0; gsel < 2; ++gsel) {
        const int g = kk * 32 + gsel * 16 + half * 8;     // 8 consecutive leaves
        float pref = 1.0f;                                // levels 0..4 shared by the run
        #pragma unroll
        for (int l = 0; l < 5; ++l) {
          const int node = ((1 << l) - 1) + (g >> (8 - l));
          const float v = (float)vr[node];
          pref *= ((g >> (7 - l)) & 1) ? v : (1.0f - v);
        }
        const float v5 = (float)vr[31 + (g >> 3)];        // level 5: shared node
        #pragma unroll
        for (int u = 0; u < 8; ++u) {
          const int j = g + u;
          float p = pref * (((j >> 2) & 1) ? v5 : (1.0f - v5));
          const float v6 = (float)vr[63 + (j >> 2)];
          p *= ((j >> 1) & 1) ? v6 : (1.0f - v6);
          const float v7 = (float)vr[127 + (j >> 1)];
          p *= (j & 1) ? v7 : (1.0f - v7);
          pa.h[gsel * 8 + u] = (__bf16)p;
        }
      }
      af[kk] = pa;                                        // reuse A-frag registers for p
    }

    // ================= phase 3: out = p @ leaf_probs =================
    for (int ct = 0; ct < 16; ++ct) {
      AccU acc;
      #pragma unroll
      for (int r = 0; r < 8; ++r) acc.f[r] = 0.0f;
      const int c = ct * 16 + m;                          // lane's class column (N)
      #pragma unroll
      for (int kk = 0; kk < 8; ++kk) {
        BFrag b;                                          // B[k][n] = leaf[k][n] -> LT row `c`
        b.q[0] = ((const uint4*)(LT + c * kPS + kk * 32 + half * 8))[0];
        b.q[1] = ((const uint4*)(LT + c * kPS + kk * 32 + 16 + half * 8))[0];
        acc.v = __builtin_amdgcn_wmma_f32_16x16x32_bf16(
            false, af[kk].v, false, b.v, (short)0, acc.v, false, false);
      }
      #pragma unroll
      for (int r = 0; r < 8; ++r)
        out[(size_t)(row0 + half * 8 + r) * kC + c] = acc.f[r];
    }

    // ================= phase 4: hard path index (one lane per row) ==========
    if (half == 0) {
      int id = 0;
      #pragma unroll
      for (int l = 0; l < 8; ++l) {
        const int node = ((1 << l) - 1) + id;
        const float v = (float)vr[node];
        id = 2 * id + (v >= 0.5f ? 1 : 0);
      }
      idx_out[row0 + m] = id;
    }
  }
}

extern "C" void kernel_launch(void* const* d_in, const int* in_sizes, int n_in,
                              void* d_out, int out_size, void* d_ws, size_t ws_size,
                              hipStream_t stream) {
  const float* x  = (const float*)d_in[0];   // [B, 256] f32
  const float* W  = (const float*)d_in[1];   // [255, 256] f32
  const float* b  = (const float*)d_in[2];   // [255] f32
  const float* L  = (const float*)d_in[3];   // [256, 256] f32
  float* out = (float*)d_out;                // [B, 256] f32 then [B] i32
  const int Btot = in_sizes[0] / kF;
  int* idx_out = (int*)(out + (size_t)Btot * kC);

  const int numTiles = Btot / kRowsPerBlock;            // B=65536 -> 1024 tiles
  const int grid = numTiles < 512 ? numTiles : 512;
  const size_t ldsBytes =
      (size_t)(256 /*W*/ + 256 /*leaf^T*/ + kWaves * 16 /*vals*/) * kPS *
      sizeof(__bf16);                                   // 304,128 B <= 320 KB/WGP

  soft_tree_fused<<<dim3(grid), dim3(kWaves * 32), ldsBytes, stream>>>(
      x, W, b, L, out, idx_out, numTiles);
}
